// OneLayerModel_21354577395867
// MI455X (gfx1250) — matrix operand order repaired and checked by reading, not compiled, not run on hardware
//
#include <hip/hip_runtime.h>
#include <hip/hip_bf16.h>

// ---------------- model constants ----------------
#define BB     64
#define TT     32
#define CC     384
#define HH     6
#define DD     64
#define NL     2
#define VV     50257
#define MM     (BB*TT)          // 2048
#define NQKV   (3*HH*DD)        // 1152
#define VPAD   50304            // V padded to multiple of 128

typedef __attribute__((ext_vector_type(16))) __bf16 v16bf;
typedef __attribute__((ext_vector_type(8)))  __bf16 v8bf;
typedef __attribute__((ext_vector_type(8)))  float  v8f;

// ---- gfx1250 async global->LDS copy (ASYNCcnt path), with safe fallback ----
#if __has_builtin(__builtin_amdgcn_global_load_async_to_lds_b128) && \
    __has_builtin(__builtin_amdgcn_s_wait_asynccnt)
#define GLT_ASYNC_COPY 1
typedef __attribute__((ext_vector_type(4))) int glt_v4i;
typedef __attribute__((address_space(1))) glt_v4i* glt_g4p;   // global int4*
typedef __attribute__((address_space(3))) glt_v4i* glt_l4p;   // LDS    int4*
#define GLT_AS1(p) ((glt_g4p)(unsigned long long)(p))
#define GLT_AS3(p) ((glt_l4p)(unsigned long long)(p))
#else
#define GLT_ASYNC_COPY 0
#endif

// ---------------- embedding + bf16 mirror ----------------
__global__ __launch_bounds__(CC)
void glt_embed_kernel(const int* __restrict__ x, const float* __restrict__ w_e,
                      const float* __restrict__ pos, float* __restrict__ h32,
                      __bf16* __restrict__ h16)
{
    int m = blockIdx.x;          // 0..2047  (b*T + t)
    int c = threadIdx.x;         // 0..383
    int tok = x[m];
    float v = w_e[(size_t)tok*CC + c] + pos[(m & (TT-1))*CC + c];
    size_t idx = (size_t)m*CC + c;
    h32[idx] = v;
    h16[idx] = (__bf16)v;
}

// ---------------- weight conversions ----------------
// packed per-layer QKV weight: wq16[l][n][k], n = sel*384 + h*64 + d
__global__ __launch_bounds__(256)
void glt_cvt_qkvw_kernel(const float* __restrict__ Wq, const float* __restrict__ Wk,
                         const float* __restrict__ Wv, __bf16* __restrict__ out)
{
    int idx = blockIdx.x*256 + threadIdx.x;
    if (idx >= NL*NQKV*CC) return;
    int l  = idx / (NQKV*CC);
    int r  = idx % (NQKV*CC);
    int n  = r / CC;
    int k  = r % CC;
    int sel = n / (HH*DD);
    int hd  = n % (HH*DD);
    const float* W = (sel == 0) ? Wq : (sel == 1) ? Wk : Wv;
    out[idx] = (__bf16)W[(size_t)l*(HH*DD*CC) + (size_t)hd*CC + k];
}

__global__ __launch_bounds__(256)
void glt_cvt_lmw_kernel(const float* __restrict__ lm_w, __bf16* __restrict__ out)
{
    size_t idx = (size_t)blockIdx.x*256 + threadIdx.x;
    if (idx >= (size_t)VPAD*CC) return;
    size_t n = idx / CC, k = idx % CC;
    out[idx] = (n < VV) ? (__bf16)lm_w[n*CC + k] : (__bf16)0.0f;
}

// ---------------- bf16 WMMA GEMM, async double-buffered ----------------
// C[m,n] = sum_k A[m,k]*Bw[n,k] (+bias[n]); A:[MxK] bf16 rm, Bw:[Npad x K] bf16 rm,
// C:[M x ldc] fp32, write only n < Nvalid.
// Block tile 128x128, 256 thr (8 waves); wave tile 32x64 = 2x4 accs of 16x16.
// K consumed in 32-wide chunks, ping/pong LDS, global->LDS via async b128 copies.
#define GEMM_BM  128
#define GEMM_BN  128
#define GEMM_KC  32
#define GEMM_ST  40   // LDS row stride in bf16 (80B: 16B-aligned, bank-spreading)
__global__ __launch_bounds__(256)
void glt_gemm_bf16_wmma(const __bf16* __restrict__ A, const __bf16* __restrict__ Bw,
                        float* __restrict__ C, const float* __restrict__ bias,
                        int M, int K, int ldc, int Nvalid)
{
    __shared__ __attribute__((aligned(16))) __bf16 As[2][GEMM_BM*GEMM_ST];
    __shared__ __attribute__((aligned(16))) __bf16 Bs[2][GEMM_BN*GEMM_ST];

    const int tid  = threadIdx.x;
    const int wave = tid >> 5;        // 0..7
    const int lane = tid & 31;
    const int g    = lane >> 4;       // lane half
    const int ln   = lane & 15;
    const int bm   = blockIdx.x * GEMM_BM;
    const int bn   = blockIdx.y * GEMM_BN;
    const int wm   = (wave >> 1) * 32;   // 0,32,64,96
    const int wn   = (wave & 1) * 64;    // 0,64

    // per-thread copy slots: 2 x uint4 per matrix per chunk
    const int r0 = tid >> 2;              // rows 0..63   (it=0)
    const int r1 = (tid + 256) >> 2;      // rows 64..127 (it=1)
    const int c8 = (tid & 3) << 3;        // 0,8,16,24 bf16 within the 32-wide chunk

    v8f acc[2][4] = {};

    const int nch = K / GEMM_KC;

    // ---- chunk issue: global -> LDS ----
    auto issue_chunk = [&](int buf, int kc) {
#if GLT_ASYNC_COPY
        __builtin_amdgcn_global_load_async_to_lds_b128(
            GLT_AS1(&A [(size_t)(bm + r0)*K + kc + c8]), GLT_AS3(&As[buf][r0*GEMM_ST + c8]), 0, 0);
        __builtin_amdgcn_global_load_async_to_lds_b128(
            GLT_AS1(&Bw[(size_t)(bn + r0)*K + kc + c8]), GLT_AS3(&Bs[buf][r0*GEMM_ST + c8]), 0, 0);
        __builtin_amdgcn_global_load_async_to_lds_b128(
            GLT_AS1(&A [(size_t)(bm + r1)*K + kc + c8]), GLT_AS3(&As[buf][r1*GEMM_ST + c8]), 0, 0);
        __builtin_amdgcn_global_load_async_to_lds_b128(
            GLT_AS1(&Bw[(size_t)(bn + r1)*K + kc + c8]), GLT_AS3(&Bs[buf][r1*GEMM_ST + c8]), 0, 0);
#else
        *(uint4*)&As[buf][r0*GEMM_ST + c8] = *(const uint4*)&A [(size_t)(bm + r0)*K + kc + c8];
        *(uint4*)&Bs[buf][r0*GEMM_ST + c8] = *(const uint4*)&Bw[(size_t)(bn + r0)*K + kc + c8];
        *(uint4*)&As[buf][r1*GEMM_ST + c8] = *(const uint4*)&A [(size_t)(bm + r1)*K + kc + c8];
        *(uint4*)&Bs[buf][r1*GEMM_ST + c8] = *(const uint4*)&Bw[(size_t)(bn + r1)*K + kc + c8];
#endif
    };

    issue_chunk(0, 0);

    for (int i = 0; i < nch; ++i) {
        if (i + 1 < nch) {
            issue_chunk((i + 1) & 1, (i + 1) * GEMM_KC);
#if GLT_ASYNC_COPY
            __builtin_amdgcn_s_wait_asynccnt(4);   // chunk i landed; i+1 still in flight
#endif
        } else {
#if GLT_ASYNC_COPY
            __builtin_amdgcn_s_wait_asynccnt(0);
#endif
        }
        __syncthreads();

        const int buf = i & 1;
        v16bf afrag[2], bfrag[4];
        // A 16x32 frag: lane row (wm+mi*16+ln), K = {g*8+0..7, 16+g*8+0..7}
        #pragma unroll
        for (int mi = 0; mi < 2; ++mi) {
            const __bf16* base = &As[buf][(wm + mi*16 + ln)*GEMM_ST + g*8];
            v8bf lo = *(const v8bf*)base;
            v8bf hi = *(const v8bf*)(base + 16);
            #pragma unroll
            for (int e = 0; e < 8; ++e) { afrag[mi][e] = lo[e]; afrag[mi][8+e] = hi[e]; }
        }
        // B 32x16 frag: lane col (wn+ni*16+ln), K = {g*8+0..7, 16+g*8+0..7}
        #pragma unroll
        for (int ni = 0; ni < 4; ++ni) {
            const __bf16* base = &Bs[buf][(wn + ni*16 + ln)*GEMM_ST + g*8];
            v8bf lo = *(const v8bf*)base;
            v8bf hi = *(const v8bf*)(base + 16);
            #pragma unroll
            for (int e = 0; e < 8; ++e) { bfrag[ni][e] = lo[e]; bfrag[ni][8+e] = hi[e]; }
        }
        #pragma unroll
        for (int mi = 0; mi < 2; ++mi)
            #pragma unroll
            for (int ni = 0; ni < 4; ++ni)
                acc[mi][ni] = __builtin_amdgcn_wmma_f32_16x16x32_bf16(
                    false, afrag[mi], false, bfrag[ni],
                    (short)0, acc[mi][ni], false, false);
        __syncthreads();
    }

    // C/D layout: VGPR r -> row r + 8*g, col = ln
    #pragma unroll
    for (int ni = 0; ni < 4; ++ni) {
        int n = bn + wn + ni*16 + ln;
        if (n < Nvalid) {
            float bv = bias ? bias[n] : 0.0f;
            #pragma unroll
            for (int mi = 0; mi < 2; ++mi)
                #pragma unroll
                for (int r = 0; r < 8; ++r) {
                    int m = bm + wm + mi*16 + r + g*8;
                    C[(size_t)m*ldc + n] = acc[mi][ni][r] + bv;
                }
        }
    }
}

// ---------------- fused per-(b,head) causal attention + residual ----------------
__global__ __launch_bounds__(256)
void glt_attn_kernel(const float* __restrict__ qkv, float* __restrict__ h32,
                     __bf16* __restrict__ h16)
{
    __shared__ float qs[TT][DD], ks[TT][DD], vs[TT][DD];
    __shared__ float wei[TT][TT + 1];
    int b  = blockIdx.x / HH;
    int hh = blockIdx.x % HH;
    int tid = threadIdx.x;

    for (int i = tid; i < TT*DD; i += 256) {
        int t = i >> 6, d = i & 63;
        size_t base = (size_t)(b*TT + t)*NQKV + hh*DD + d;
        qs[t][d] = qkv[base];
        ks[t][d] = qkv[base + HH*DD];
        vs[t][d] = qkv[base + 2*HH*DD];
    }
    __syncthreads();

    for (int i = tid; i < TT*TT; i += 256) {
        int t = i >> 5, s = i & 31;
        float a = -3.0e38f;
        if (s <= t) {
            a = 0.0f;
            #pragma unroll 8
            for (int d = 0; d < DD; ++d) a += qs[t][d] * ks[s][d];
            a *= 0.125f;   // 1/sqrt(64)
        }
        wei[t][s] = a;
    }
    __syncthreads();

    if (tid < TT) {
        int t = tid;
        float mx = -3.0e38f;
        for (int s = 0; s <= t; ++s) mx = fmaxf(mx, wei[t][s]);
        float sum = 0.0f;
        for (int s = 0; s < TT; ++s) {
            float e = (s <= t) ? __expf(wei[t][s] - mx) : 0.0f;
            wei[t][s] = e;
            sum += e;
        }
        float inv = 1.0f / sum;
        for (int s = 0; s < TT; ++s) wei[t][s] *= inv;
    }
    __syncthreads();

    for (int i = tid; i < TT*DD; i += 256) {
        int t = i >> 6, d = i & 63;
        float a = 0.0f;
        for (int s = 0; s <= t; ++s) a += wei[t][s] * vs[s][d];
        size_t hidx = (size_t)(b*TT + t)*CC + hh*DD + d;   // head owns cols [h*64, h*64+64)
        float nv = h32[hidx] + a;
        h32[hidx] = nv;
        h16[hidx] = (__bf16)nv;
    }
}

// ---------------- single-pass online row max / sum(exp) over vocab ----------------
__global__ __launch_bounds__(256)
void glt_rowstats_kernel(const float* __restrict__ logits,
                         float* __restrict__ rowmax, float* __restrict__ rowsum)
{
    __shared__ float smax[256], ssum[256];
    int m = blockIdx.x, tid = threadIdx.x;
    const float* row = logits + (size_t)m*VV;
    float mx = -3.0e38f, sm = 0.0f;
    for (int n = tid; n < VV; n += 256) {
        float v = row[n];
        if (v > mx) { sm *= __expf(mx - v); mx = v; }
        sm += __expf(v - mx);
    }
    smax[tid] = mx; ssum[tid] = sm; __syncthreads();
    for (int s = 128; s > 0; s >>= 1) {
        if (tid < s) {
            float m1 = smax[tid], m2 = smax[tid + s];
            float s1 = ssum[tid], s2 = ssum[tid + s];
            float mo = fmaxf(m1, m2);
            smax[tid] = mo;
            ssum[tid] = s1*__expf(m1 - mo) + s2*__expf(m2 - mo);
        }
        __syncthreads();
    }
    if (tid == 0) { rowmax[m] = smax[0]; rowsum[m] = ssum[0]; }
}

// ---------------- mean cross-entropy ----------------
__global__ __launch_bounds__(256)
void glt_loss_kernel(const float* __restrict__ logits, const float* __restrict__ rowmax,
                     const float* __restrict__ rowsum, const int* __restrict__ targets,
                     float* __restrict__ out_loss)
{
    __shared__ float red[256];
    int tid = threadIdx.x;
    float acc = 0.0f;
    for (int m = tid; m < MM; m += 256) {
        int tgt = targets[m];
        float lp = logits[(size_t)m*VV + tgt] - rowmax[m] - __logf(rowsum[m]);
        acc -= lp;
    }
    red[tid] = acc; __syncthreads();
    for (int s = 128; s > 0; s >>= 1) {
        if (tid < s) red[tid] += red[tid + s];
        __syncthreads();
    }
    if (tid == 0) out_loss[0] = red[0] / (float)MM;
}

// ---------------- host-side orchestration ----------------
static inline size_t glt_align256(size_t v) { return (v + 255) & ~(size_t)255; }

extern "C" void kernel_launch(void* const* d_in, const int* in_sizes, int n_in,
                              void* d_out, int out_size, void* d_ws, size_t ws_size,
                              hipStream_t stream) {
    (void)in_sizes; (void)n_in; (void)out_size; (void)ws_size;
    const int*   x       = (const int*)  d_in[0];
    const int*   targets = (const int*)  d_in[1];
    const float* w_e     = (const float*)d_in[2];
    const float* pos_emb = (const float*)d_in[3];
    const float* Wq      = (const float*)d_in[4];
    const float* Wk      = (const float*)d_in[5];
    const float* Wv      = (const float*)d_in[6];
    const float* lm_w    = (const float*)d_in[7];
    const float* lm_b    = (const float*)d_in[8];
    float* logits = (float*)d_out;                       // [2048 x 50257]
    float* loss   = (float*)d_out + (size_t)MM*VV;       // 1 scalar after logits

    // workspace carve-up
    char* ws = (char*)d_ws;
    size_t off = 0;
    float*  h32    = (float*)(ws + off);  off = glt_align256(off + (size_t)MM*CC*4);
    __bf16* h16    = (__bf16*)(ws + off); off = glt_align256(off + (size_t)MM*CC*2);
    float*  qkv    = (float*)(ws + off);  off = glt_align256(off + (size_t)MM*NQKV*4);
    __bf16* wq16   = (__bf16*)(ws + off); off = glt_align256(off + (size_t)NL*NQKV*CC*2);
    __bf16* lmw16  = (__bf16*)(ws + off); off = glt_align256(off + (size_t)VPAD*CC*2);
    float*  rowmax = (float*)(ws + off);  off = glt_align256(off + (size_t)MM*4);
    float*  rowsum = (float*)(ws + off);  off = glt_align256(off + (size_t)MM*4);

    // 1. embedding
    glt_embed_kernel<<<MM, CC, 0, stream>>>(x, w_e, pos_emb, h32, h16);

    // 2. weight conversions
    {
        int n = NL*NQKV*CC;
        glt_cvt_qkvw_kernel<<<(n + 255)/256, 256, 0, stream>>>(Wq, Wk, Wv, wq16);
        size_t n2 = (size_t)VPAD*CC;
        glt_cvt_lmw_kernel<<<(int)((n2 + 255)/256), 256, 0, stream>>>(lm_w, lmw16);
    }

    // 3. transformer layers: QKV WMMA GEMM + fused attention/residual
    for (int l = 0; l < NL; ++l) {
        dim3 grid(MM/GEMM_BM, NQKV/GEMM_BN);   // 16 x 9
        glt_gemm_bf16_wmma<<<grid, 256, 0, stream>>>(
            h16, wq16 + (size_t)l*NQKV*CC, qkv, nullptr,
            MM, CC, NQKV, NQKV);
        glt_attn_kernel<<<BB*HH, 256, 0, stream>>>(qkv, h32, h16);
    }

    // 4. logits = h @ lm_w^T + lm_b  (dominant 79-GFLOP WMMA GEMM)
    {
        dim3 grid(MM/GEMM_BM, VPAD/GEMM_BN);   // 16 x 393
        glt_gemm_bf16_wmma<<<grid, 256, 0, stream>>>(
            h16, lmw16, logits, lm_b, MM, CC, VV, VV);
    }

    // 5. log-softmax stats + loss
    glt_rowstats_kernel<<<MM, 256, 0, stream>>>(logits, rowmax, rowsum);
    glt_loss_kernel<<<1, 256, 0, stream>>>(logits, rowmax, rowsum, targets, loss);
}
